// MultiHeadAttention_10170482557219
// MI455X (gfx1250) — matrix operand order repaired and checked by reading, not compiled, not run on hardware
//
#include <hip/hip_runtime.h>

// ---------------------------------------------------------------------------
// MHA forward for MI455X (gfx1250, wave32, WMMA + TDM).
// Roofline: ~69 GFLOP vs ~100MB HBM traffic @23.3TB/s -> compute bound;
// all GEMMs routed through v_wmma_f32_16x16x32_f16 (fp16 in / fp32 acc).
// Out-proj stages its fp16 A tile with tensor_load_to_lds (TDM) using the
// hardware pad feature to reproduce the padded LDS layout (LSTR=40 halves).
// ---------------------------------------------------------------------------

#define DEV __device__ __forceinline__

typedef _Float16 v16h __attribute__((ext_vector_type(16)));
typedef _Float16 v8h  __attribute__((ext_vector_type(8)));
typedef float    v8f  __attribute__((ext_vector_type(8)));
typedef unsigned int u32x4 __attribute__((ext_vector_type(4)));
typedef int          i32x4 __attribute__((ext_vector_type(4)));
typedef int          i32x8 __attribute__((ext_vector_type(8)));

static constexpr int   B_  = 2;
static constexpr int   S_  = 2048;
static constexpr int   E_  = 1024;
static constexpr int   H_  = 16;
static constexpr int   D_  = 64;
static constexpr float SCALE_  = 0.125f;                 // 1/sqrt(64)
static constexpr float ROPE_C_ = 0.2878231366242557f;    // ln(10000)/32
static constexpr int   LSTR = 40;                        // LDS row stride (halves)

#if __has_builtin(__builtin_amdgcn_tensor_load_to_lds)
#define HAVE_TDM 1
#endif

DEV v8f vzero() {
  v8f z;
#pragma unroll
  for (int i = 0; i < 8; ++i) z[i] = 0.0f;
  return z;
}

DEV v16h vcat(v8h lo, v8h hi) {
  v16h r;
#pragma unroll
  for (int i = 0; i < 8; ++i) { r[i] = lo[i]; r[i + 8] = hi[i]; }
  return r;
}

DEV v8f wmma_f16(v16h a, v16h b, v8f c) {
  return __builtin_amdgcn_wmma_f32_16x16x32_f16(false, a, false, b, (short)0, c,
                                                false, false);
}

// A fragment (16x32 f16, M x K): lane<16 -> K 0-15, lane>=16 -> K 16-31.
DEV v16h a_frag_row(const _Float16* rowp, int lane) {
  const int off = (lane & 16) ? 16 : 0;
  v8h lo = *(const v8h*)(rowp + off);
  v8h hi = *(const v8h*)(rowp + 8 + off);
  return vcat(lo, hi);
}

// B fragment (32x16 f16, K x N): lane&15 -> N, lane<16 -> K 0-15, lane>=16 -> K 16-31.
DEV v16h b_frag_row(const _Float16* rowp, int lane) {
  const _Float16* p = rowp + ((lane & 16) ? 16 : 0);
  v8h lo = *(const v8h*)p;
  v8h hi = *(const v8h*)(p + 8);
  return vcat(lo, hi);
}

// 16-lane-group reductions (rows of a C-tile live in a half-wave group).
DEV float hmax16(float v) {
#pragma unroll
  for (int m = 1; m < 16; m <<= 1) v = fmaxf(v, __shfl_xor(v, m, 32));
  return v;
}
DEV float hsum16(float v) {
#pragma unroll
  for (int m = 1; m < 16; m <<= 1) v += __shfl_xor(v, m, 32);
  return v;
}

#ifdef HAVE_TDM
// Issue a TDM 2-D tile load: tile_dim0 x tile_dim1 elements of 2 bytes from a
// row-major global tensor (row stride = gstride elements) into LDS at laddr,
// padding 4 DWORDs after every 16 DWORDs (i.e. 32-half rows -> LSTR=40).
DEV void tdm_load_tile_2d_f16(const _Float16* gptr, unsigned laddr,
                              unsigned gstride, unsigned rows) {
  const unsigned long long ga = (unsigned long long)(uintptr_t)gptr;
  u32x4 g0;
  g0[0] = 1u;                                            // count=1 (valid user D#)
  g0[1] = laddr;                                         // lds_addr (bytes)
  g0[2] = (unsigned)ga;                                  // global_addr[31:0]
  g0[3] = (unsigned)((ga >> 32) & 0x1FFFFFFu) | (2u << 30);  // addr[56:32] | type=2
  i32x8 g1;
  g1[0] = (int)((1u << 16)      // data_size = 2 bytes
              | (1u << 20)      // pad_enable
              | (3u << 22)      // pad_interval: 16 DWORDs
              | (3u << 25));    // pad_amount: 4 DWORDs (8 halves)
  g1[1] = (int)((gstride & 0xFFFFu) << 16);              // tensor_dim0[15:0] @ bits 63:48
  g1[2] = (int)(((gstride >> 16) & 0xFFFFu)              // tensor_dim0[31:16]
              | ((rows & 0xFFFFu) << 16));               // tensor_dim1[15:0]
  g1[3] = (int)(((rows >> 16) & 0xFFFFu)                 // tensor_dim1[31:16]
              | (32u << 16));                            // tile_dim0 = 32
  g1[4] = (int)rows;                                     // tile_dim1 = rows; tile_dim2 = 0
  g1[5] = (int)gstride;                                  // tensor_dim0_stride[31:0]
  g1[6] = 0;                                             // stride hi / dim1_stride lo
  g1[7] = 0;
  i32x4 gz;
  gz[0] = gz[1] = gz[2] = gz[3] = 0;
#if __clang_major__ >= 23
  i32x8 gz8;
#pragma unroll
  for (int i = 0; i < 8; ++i) gz8[i] = 0;
  __builtin_amdgcn_tensor_load_to_lds(g0, g1, gz, gz, gz8, 0);
#else
  __builtin_amdgcn_tensor_load_to_lds(g0, g1, gz, gz, 0);
#endif
}
#endif  // HAVE_TDM

// ---------------------------------------------------------------------------
// QKV projection: C = X @ W^T + b, fused RoPE (Q/K) / transpose (V), fp16 out.
// MODE 0: rope, store [B,H,S,D] (Q and K).  MODE 1: no rope, store [B,H,D,S] (V).
// Block: 128x128 tile, 8 waves, each wave = 16(M) x 128(N), K-step 32.
// ---------------------------------------------------------------------------
template <int MODE>
__global__ void __launch_bounds__(256)
proj_qkv_kernel(const float* __restrict__ X, const float* __restrict__ W,
                const float* __restrict__ bias, _Float16* __restrict__ out) {
  __shared__ __align__(32) _Float16 Xs[128 * LSTR];
  __shared__ __align__(32) _Float16 Ws[128 * LSTR];

  const int lane = threadIdx.x & 31;
  const int wv   = threadIdx.x >> 5;
  const int bn   = blockIdx.x;  // N / 128
  const int bm   = blockIdx.y;  // M / 128

  const int ldr = threadIdx.x >> 1;         // staging row 0..127
  const int ldc = (threadIdx.x & 1) * 16;   // staging col 0 / 16

  v8f acc[8];
#pragma unroll
  for (int f = 0; f < 8; ++f) acc[f] = vzero();

  const int arow = 16 * wv + (lane & 15);

  for (int kb = 0; kb < E_; kb += 32) {
    // ---- stage X tile and W tile (fp32 -> fp16) into LDS --------------------
    const float4* xp = (const float4*)(X + (size_t)(bm * 128 + ldr) * E_ + kb + ldc);
    const float4* wp = (const float4*)(W + (size_t)(bn * 128 + ldr) * E_ + kb + ldc);
    float fx[16], fw[16];
#pragma unroll
    for (int q = 0; q < 4; ++q) {
      float4 a = xp[q]; float4 b = wp[q];
      fx[4 * q + 0] = a.x; fx[4 * q + 1] = a.y; fx[4 * q + 2] = a.z; fx[4 * q + 3] = a.w;
      fw[4 * q + 0] = b.x; fw[4 * q + 1] = b.y; fw[4 * q + 2] = b.z; fw[4 * q + 3] = b.w;
    }
    v8h hx0, hx1, hw0, hw1;
#pragma unroll
    for (int i = 0; i < 8; ++i) {
      hx0[i] = (_Float16)fx[i]; hx1[i] = (_Float16)fx[i + 8];
      hw0[i] = (_Float16)fw[i]; hw1[i] = (_Float16)fw[i + 8];
    }
    _Float16* xs = Xs + ldr * LSTR + ldc;
    _Float16* ws = Ws + ldr * LSTR + ldc;
    *(v8h*)xs = hx0; *(v8h*)(xs + 8) = hx1;
    *(v8h*)ws = hw0; *(v8h*)(ws + 8) = hw1;
    __syncthreads();

    // ---- 8 WMMAs per K-step per wave ---------------------------------------
    v16h a = a_frag_row(Xs + arow * LSTR, lane);
#pragma unroll
    for (int f = 0; f < 8; ++f) {
      v16h b = b_frag_row(Ws + (16 * f + (lane & 15)) * LSTR, lane);
      acc[f] = wmma_f16(a, b, acc[f]);
    }
    __syncthreads();
  }

  // ---- epilogue: bias (+RoPE) + fp16 store ---------------------------------
  const int hi8 = (lane >= 16) ? 8 : 0;
  const int n0  = bn * 128 + (lane & 15);
#pragma unroll
  for (int f = 0; f < 8; ++f) {
    const int   n  = n0 + 16 * f;
    const float bv = bias[n];
#pragma unroll
    for (int j = 0; j < 8; ++j) {
      const int m = bm * 128 + 16 * wv + j + hi8;
      float x = acc[f][j] + bv;
      const int d = n & (D_ - 1);
      if (MODE == 0) {  // RoPE (interleaved rotate, concatenated cos/sin tables)
        const int   srow = m & (S_ - 1);
        const float xo   = __shfl_xor(x, 1, 32);        // neighbor column n^1
        const float fr   = __expf(-(float)(d & 31) * ROPE_C_);
        const float th   = (float)srow * fr;
        float cs, sn;
        __sincosf(th, &sn, &cs);
        const float x2 = (d & 1) ? xo : -xo;
        x = x * cs + x2 * sn;
      }
      const int b = m >> 11;
      const int srow = m & (S_ - 1);
      const int h = n >> 6;
      if (MODE == 1)
        out[((size_t)(b * H_ + h) * D_ + d) * S_ + srow] = (_Float16)x;  // V^T
      else
        out[((size_t)(b * H_ + h) * S_ + srow) * D_ + d] = (_Float16)x;  // Q/K
    }
  }
}

// ---------------------------------------------------------------------------
// Flash attention: one wave per 16-row Q tile; 32 keys per iteration.
// scores (2 C-tiles, 4 WMMAs) -> online softmax -> P (via LDS C->A relayout)
// -> P@V (4 WMMAs into 16x64 accumulator).
// ---------------------------------------------------------------------------
__global__ void __launch_bounds__(256)
attn_kernel(const _Float16* __restrict__ Q, const _Float16* __restrict__ K,
            const _Float16* __restrict__ Vt, const int* __restrict__ mask,
            _Float16* __restrict__ O) {
  __shared__ __align__(32) _Float16 P[8][16 * LSTR];

  const int lane = threadIdx.x & 31;
  const int wv   = threadIdx.x >> 5;
  const int qt   = blockIdx.x * 8 + wv;   // 0..4095
  const int bh   = qt >> 7;               // (b,h) pair
  const int qi   = qt & 127;
  const int b    = bh >> 4;
  const int h    = bh & 15;

  const _Float16* Qp = Q  + (size_t)bh * S_ * D_;
  const _Float16* Kp = K  + (size_t)bh * S_ * D_;
  const _Float16* Vp = Vt + (size_t)bh * D_ * S_;
  const int*      mp = mask + b * S_;

  const int q0   = qi * 16;
  const int col  = lane & 15;             // C-layout N index / frag row select
  const int hi16 = (lane >= 16);
  const int hi8  = hi16 ? 8 : 0;

  const _Float16* qrow = Qp + (size_t)(q0 + col) * D_;
  const v16h aq0 = a_frag_row(qrow, lane);       // d 0..31
  const v16h aq1 = a_frag_row(qrow + 32, lane);  // d 32..63

  v8f o[4];
#pragma unroll
  for (int t = 0; t < 4; ++t) o[t] = vzero();
  float rm[8], rs[8];
#pragma unroll
  for (int j = 0; j < 8; ++j) { rm[j] = -1e30f; rs[j] = 0.0f; }

  _Float16* Pl = &P[wv][0];

  for (int kt = 0; kt < S_; kt += 32) {
    if (kt + 32 < S_)  // pull next K tile toward the WGP (global_prefetch_b8)
      __builtin_prefetch(Kp + (size_t)(kt + 32 + col) * D_, 0, 1);

    // ---- scores: two 16x16 C-tiles over 32 keys ----------------------------
    v8f sc[2];
#pragma unroll
    for (int c = 0; c < 2; ++c) {
      const _Float16* kp = Kp + (size_t)(kt + 16 * c + col) * D_ + (hi16 ? 16 : 0);
      v16h kb0 = *(const v16h*)kp;          // d 0..31 chunk (B layout)
      v16h kb1 = *(const v16h*)(kp + 32);   // d 32..63 chunk
      v8f  z   = vzero();
      z = wmma_f16(aq0, kb0, z);
      sc[c] = wmma_f16(aq1, kb1, z);
    }

    const int m0 = mp[kt + col];
    const int m1 = mp[kt + 16 + col];

    // ---- online softmax ----------------------------------------------------
    float tm[8];
#pragma unroll
    for (int j = 0; j < 8; ++j) {
      float s0 = m0 ? sc[0][j] * SCALE_ : -1e30f;
      float s1 = m1 ? sc[1][j] * SCALE_ : -1e30f;
      sc[0][j] = s0; sc[1][j] = s1;
      tm[j] = hmax16(fmaxf(s0, s1));
    }
#pragma unroll
    for (int j = 0; j < 8; ++j) {
      const float nm    = fmaxf(rm[j], tm[j]);
      const float alpha = __expf(rm[j] - nm);
      const float p0    = __expf(sc[0][j] - nm);
      const float p1    = __expf(sc[1][j] - nm);
      rs[j] = rs[j] * alpha + hsum16(p0 + p1);
      rm[j] = nm;
#pragma unroll
      for (int t = 0; t < 4; ++t) o[t][j] *= alpha;
      const int rb = (j + hi8) * LSTR;           // C->A relayout via LDS
      Pl[rb + col]      = (_Float16)p0;
      Pl[rb + 16 + col] = (_Float16)p1;
    }

    // ---- P @ V -------------------------------------------------------------
    const v16h pa = a_frag_row(Pl + col * LSTR, lane);
#pragma unroll
    for (int t = 0; t < 4; ++t) {
      const _Float16* vp = Vp + (size_t)(16 * t + col) * S_ + kt + (hi16 ? 16 : 0);
      v16h vb = *(const v16h*)vp;
      o[t] = wmma_f16(pa, vb, o[t]);
    }
  }

  // ---- normalize + store to [B*S, E] fp16 workspace ------------------------
  float inv[8];
#pragma unroll
  for (int j = 0; j < 8; ++j) inv[j] = (rs[j] > 0.0f) ? 1.0f / rs[j] : 0.0f;
#pragma unroll
  for (int t = 0; t < 4; ++t) {
    const int e = h * D_ + 16 * t + col;
#pragma unroll
    for (int j = 0; j < 8; ++j) {
      const int qr = q0 + j + hi8;
      O[((size_t)(b * S_ + qr)) * E_ + e] = (_Float16)(o[t][j] * inv[j]);
    }
  }
}

// ---------------------------------------------------------------------------
// Output projection: out = A(fp16) @ Wo^T + bo, fp32 result.
// A tile staged by the Tensor Data Mover (pad feature reproduces LSTR layout);
// W tile staged by threads (needs fp32 -> fp16 conversion, TDM moves raw bytes).
// ---------------------------------------------------------------------------
__global__ void __launch_bounds__(256)
out_proj_kernel(const _Float16* __restrict__ Xh, const float* __restrict__ W,
                const float* __restrict__ bias, float* __restrict__ out) {
  __shared__ __align__(32) _Float16 Xs[128 * LSTR];
  __shared__ __align__(32) _Float16 Ws[128 * LSTR];

  const int lane = threadIdx.x & 31;
  const int wv   = threadIdx.x >> 5;
  const int bn   = blockIdx.x;
  const int bm   = blockIdx.y;
  const int ldr  = threadIdx.x >> 1;
  const int ldc  = (threadIdx.x & 1) * 16;

  v8f acc[8];
#pragma unroll
  for (int f = 0; f < 8; ++f) acc[f] = vzero();
  const int arow = 16 * wv + (lane & 15);

#ifdef HAVE_TDM
  const unsigned xs_lds = (unsigned)(uintptr_t)(&Xs[0]);  // LDS byte address = addr[31:0]
#endif

  for (int kb = 0; kb < E_; kb += 32) {
    // ---- W tile: fp32 -> fp16 by all threads --------------------------------
    const float4* wp = (const float4*)(W + (size_t)(bn * 128 + ldr) * E_ + kb + ldc);
    float fw[16];
#pragma unroll
    for (int q = 0; q < 4; ++q) {
      float4 w4 = wp[q];
      fw[4 * q + 0] = w4.x; fw[4 * q + 1] = w4.y; fw[4 * q + 2] = w4.z; fw[4 * q + 3] = w4.w;
    }
    v8h hw0, hw1;
#pragma unroll
    for (int i = 0; i < 8; ++i) { hw0[i] = (_Float16)fw[i]; hw1[i] = (_Float16)fw[i + 8]; }
    _Float16* ws = Ws + ldr * LSTR + ldc;
    *(v8h*)ws = hw0; *(v8h*)(ws + 8) = hw1;

    // ---- X tile (already fp16): TDM async DMA into padded LDS ---------------
#ifdef HAVE_TDM
    if (wv == 0) {
      tdm_load_tile_2d_f16(Xh + (size_t)(bm * 128) * E_ + kb, xs_lds,
                           (unsigned)E_, 128u);
      __builtin_amdgcn_s_wait_tensorcnt(0);  // issuing wave's TENSORcnt
    }
#else
    {
      const _Float16* xp = Xh + (size_t)(bm * 128 + ldr) * E_ + kb + ldc;
      v8h hx0 = *(const v8h*)xp;
      v8h hx1 = *(const v8h*)(xp + 8);
      _Float16* xs = Xs + ldr * LSTR + ldc;
      *(v8h*)xs = hx0; *(v8h*)(xs + 8) = hx1;
    }
#endif
    __syncthreads();

    v16h a = a_frag_row(Xs + arow * LSTR, lane);
#pragma unroll
    for (int f = 0; f < 8; ++f) {
      v16h bfr = b_frag_row(Ws + (16 * f + (lane & 15)) * LSTR, lane);
      acc[f] = wmma_f16(a, bfr, acc[f]);
    }
    __syncthreads();
  }

  const int hi8 = (lane >= 16) ? 8 : 0;
  const int n0  = bn * 128 + (lane & 15);
#pragma unroll
  for (int f = 0; f < 8; ++f) {
    const int   n  = n0 + 16 * f;
    const float bv = bias[n];
#pragma unroll
    for (int j = 0; j < 8; ++j) {
      const int m = bm * 128 + 16 * wv + j + hi8;
      out[(size_t)m * E_ + n] = acc[f][j] + bv;
    }
  }
}

// ---------------------------------------------------------------------------
extern "C" void kernel_launch(void* const* d_in, const int* in_sizes, int n_in,
                              void* d_out, int out_size, void* d_ws, size_t ws_size,
                              hipStream_t stream) {
  const float* q    = (const float*)d_in[0];
  const float* k    = (const float*)d_in[1];
  const float* v    = (const float*)d_in[2];
  const int*   mask = (const int*)d_in[3];
  const float* Wq = (const float*)d_in[4];  const float* bq = (const float*)d_in[5];
  const float* Wk = (const float*)d_in[6];  const float* bk = (const float*)d_in[7];
  const float* Wv = (const float*)d_in[8];  const float* bv = (const float*)d_in[9];
  const float* Wo = (const float*)d_in[10]; const float* bo = (const float*)d_in[11];
  float* out = (float*)d_out;

  char*        ws = (char*)d_ws;
  const size_t sz = (size_t)B_ * H_ * S_ * D_ * sizeof(_Float16);  // 8 MB each
  _Float16* Qw = (_Float16*)(ws);
  _Float16* Kw = (_Float16*)(ws + sz);
  _Float16* Vw = (_Float16*)(ws + 2 * sz);
  _Float16* Ow = (_Float16*)(ws + 3 * sz);

  dim3 gproj(E_ / 128, (B_ * S_) / 128);  // (8, 32)
  proj_qkv_kernel<0><<<gproj, 256, 0, stream>>>(q, Wq, bq, Qw);
  proj_qkv_kernel<0><<<gproj, 256, 0, stream>>>(k, Wk, bk, Kw);
  proj_qkv_kernel<1><<<gproj, 256, 0, stream>>>(v, Wv, bv, Vw);
  attn_kernel<<<(B_ * H_ * S_ / 16) / 8, 256, 0, stream>>>(Qw, Kw, Vw, mask, Ow);
  out_proj_kernel<<<gproj, 256, 0, stream>>>(Ow, Wo, bo, out);
}
